// MultiHeadedAttention_14310831031023
// MI455X (gfx1250) — matrix operand order repaired and checked by reading, not compiled, not run on hardware
//
#include <hip/hip_runtime.h>
#include <hip/hip_bf16.h>

typedef __attribute__((ext_vector_type(16))) _Float16 v16h;
typedef __attribute__((ext_vector_type(8)))  _Float16 v8h_t;
typedef __attribute__((ext_vector_type(4)))  _Float16 v4h_t;
typedef __attribute__((ext_vector_type(8)))  float    v8f;
typedef unsigned int u32x4 __attribute__((ext_vector_type(4)));
typedef int          i32x4 __attribute__((ext_vector_type(4)));
typedef int          i32x8 __attribute__((ext_vector_type(8)));

#define D_MODEL   512
#define NUM_HEADS 8
#define HEAD_DIM  64
#define SEQ_N     2048
#define BATCH     4

#if __has_include(<hip/amd_detail/amd_gfx1250_TDM.h>)
#define TDM_SIX_ARGS 1
#else
#define TDM_SIX_ARGS 0
#endif

// ---------------------------------------------------------------------------
// CDNA5 WMMA fragment layouts (ISA 7.12.2, wave32):
//   A (16x32 f16): lane row M = lane&15; two contiguous 8-half runs at
//       K = half*8 and K = 16 + half*8
//   B (32x16 f16): lane col N = lane&15; two contiguous 8-half runs at
//       K = half*8 and K = 16 + half*8
//   C/D (16x16 f32): lane col N = lane&15; element r -> M = r + half*8
// ---------------------------------------------------------------------------
__device__ __forceinline__ v16h concat8(v8h_t lo, v8h_t hi) {
    return __builtin_shufflevector(lo, hi, 0, 1, 2, 3, 4, 5, 6, 7,
                                           8, 9, 10, 11, 12, 13, 14, 15);
}
__device__ __forceinline__ v16h load_a_frag(const _Float16* rowPtr, int kBase, int half) {
    v8h_t lo = *(const v8h_t*)(rowPtr + kBase + half * 8);
    v8h_t hi = *(const v8h_t*)(rowPtr + kBase + 16 + half * 8);
    return concat8(lo, hi);
}
__device__ __forceinline__ v16h load_b_frag(const _Float16* rowPtr, int kBase, int half) {
    v8h_t lo = *(const v8h_t*)(rowPtr + kBase + half * 8);
    v8h_t hi = *(const v8h_t*)(rowPtr + kBase + 16 + half * 8);
    return concat8(lo, hi);
}

// ---------------------------------------------------------------------------
// Tensor Data Mover: 2D f16 tile load Global->LDS (ISA ch.8 D# layout).
// tensor row stride d0_stride (elements); LDS gets +pad DWORDs per tile row.
// ---------------------------------------------------------------------------
__device__ __forceinline__ void tdm_load_2d_f16(
    unsigned lds_addr, const void* gaddr,
    unsigned tensor_d0, unsigned tensor_d1,
    unsigned tile_d0, unsigned tile_d1,
    unsigned long long d0_stride,
    unsigned pad_interval_code, unsigned pad_amount_code)
{
    unsigned long long ga = (unsigned long long)gaddr;
    u32x4 g0;
    g0[0] = 1u;                                           // count=1 user desc
    g0[1] = lds_addr;                                     // lds_addr [63:32]
    g0[2] = (unsigned)(ga & 0xffffffffu);                 // global_addr lo
    g0[3] = (unsigned)((ga >> 32) & 0x01ffffffu) | (2u << 30);  // hi + type=2
    i32x8 g1;
    g1[0] = (int)((1u << 16) |                            // data_size = 2B
                  (1u << 20) |                            // pad_enable
                  (pad_interval_code << 22) |
                  (pad_amount_code << 25));
    g1[1] = (int)((tensor_d0 & 0xffffu) << 16);           // dim0[15:0] @ bits63:48
    g1[2] = (int)(((tensor_d0 >> 16) & 0xffffu) |
                  ((tensor_d1 & 0xffffu) << 16));         // dim0 hi | dim1 lo
    g1[3] = (int)(((tensor_d1 >> 16) & 0xffffu) |
                  ((tile_d0 & 0xffffu) << 16));           // dim1 hi | tile_dim0
    g1[4] = (int)(tile_d1 & 0xffffu);                     // tile_dim1 (dim2=0)
    g1[5] = (int)(d0_stride & 0xffffffffu);               // dim0_stride lo32
    g1[6] = (int)((d0_stride >> 32) & 0xffffu);           // dim0_stride hi16
    g1[7] = 0;
    i32x4 z4 = {0, 0, 0, 0};
#if TDM_SIX_ARGS
    i32x8 z8 = {0, 0, 0, 0, 0, 0, 0, 0};
    __builtin_amdgcn_tensor_load_to_lds(g0, g1, z4, z4, z8, 0);
#else
    __builtin_amdgcn_tensor_load_to_lds(g0, g1, z4, z4, 0);
#endif
}

// ---------------------------------------------------------------------------
// GEMM: Y[b][o][n] = sum_c W[o][c] * X[b][c][n] + bias[o]
// Block tile 128(M) x 64(N), K-step 32; 8 waves as 4x2, each 32x32.
// ---------------------------------------------------------------------------
template <typename TIN, typename TOUT>
__global__ __launch_bounds__(256)
void gemm_proj(const float* __restrict__ W, const float* __restrict__ bias,
               const TIN* __restrict__ X, TOUT* __restrict__ Y)
{
    constexpr int BM = 128, BN = 64, BK = 32, LDW = BK + 8;
    __shared__ __align__(16) _Float16 Wt[BM][LDW];
    __shared__ __align__(16) _Float16 XtT[BN][LDW];

    const int tid   = threadIdx.x;
    const int lane  = tid & 31;
    const int wave  = tid >> 5;
    const int wm    = wave >> 1;
    const int wn    = wave & 1;
    const int row16 = lane & 15;
    const int half  = lane >> 4;

    const int b  = blockIdx.z;
    const int n0 = blockIdx.x * BN;
    const int m0 = blockIdx.y * BM;

    const TIN* Xb = X + (size_t)b * D_MODEL * SEQ_N;
    TOUT*      Yb = Y + (size_t)b * D_MODEL * SEQ_N;

    v8f acc[2][2];
#pragma unroll
    for (int i = 0; i < 2; ++i)
#pragma unroll
        for (int j = 0; j < 2; ++j)
#pragma unroll
            for (int e = 0; e < 8; ++e) acc[i][j][e] = 0.0f;

    for (int k0 = 0; k0 < D_MODEL; k0 += BK) {
        // W tile (128x32) fp32 -> f16, float4 reads, v4h stores
#pragma unroll
        for (int it = 0; it < 4; ++it) {
            int q  = it * 256 + tid;
            int r  = q >> 3;
            int c4 = (q & 7) * 4;
            float4 w = *(const float4*)&W[(size_t)(m0 + r) * D_MODEL + (k0 + c4)];
            v4h_t wh;
            wh[0] = (_Float16)w.x; wh[1] = (_Float16)w.y;
            wh[2] = (_Float16)w.z; wh[3] = (_Float16)w.w;
            *(v4h_t*)&Wt[r][c4] = wh;
        }
        // X tile (32k x 64n) -> XtT[n][k]; vector global reads, scatter stores
        if constexpr (sizeof(TIN) == 4) {
#pragma unroll
            for (int it = 0; it < 2; ++it) {
                int q  = it * 256 + tid;
                int r  = q >> 4;
                int c4 = (q & 15) * 4;
                float4 x = *(const float4*)&Xb[(size_t)(k0 + r) * SEQ_N + (n0 + c4)];
                XtT[c4 + 0][r] = (_Float16)x.x;
                XtT[c4 + 1][r] = (_Float16)x.y;
                XtT[c4 + 2][r] = (_Float16)x.z;
                XtT[c4 + 3][r] = (_Float16)x.w;
            }
        } else {
            int r  = tid >> 3;
            int c8 = (tid & 7) * 8;
            v8h_t x = *(const v8h_t*)&Xb[(size_t)(k0 + r) * SEQ_N + (n0 + c8)];
#pragma unroll
            for (int e = 0; e < 8; ++e)
                XtT[c8 + e][r] = (_Float16)x[e];
        }
        __syncthreads();

        v16h af[2], bf[2];
#pragma unroll
        for (int tm = 0; tm < 2; ++tm)
            af[tm] = load_a_frag(&Wt[wm * 32 + tm * 16 + row16][0], 0, half);
#pragma unroll
        for (int tn = 0; tn < 2; ++tn)
            bf[tn] = load_b_frag(&XtT[wn * 32 + tn * 16 + row16][0], 0, half);
#pragma unroll
        for (int tm = 0; tm < 2; ++tm)
#pragma unroll
            for (int tn = 0; tn < 2; ++tn)
                acc[tm][tn] = __builtin_amdgcn_wmma_f32_16x16x32_f16(
                    false, af[tm], false, bf[tn], (short)0, acc[tm][tn],
                    false, false);
        __syncthreads();
    }

#pragma unroll
    for (int tm = 0; tm < 2; ++tm) {
#pragma unroll
        for (int tn = 0; tn < 2; ++tn) {
            const int ncol = n0 + wn * 32 + tn * 16 + row16;
#pragma unroll
            for (int r = 0; r < 8; ++r) {
                const int mrow = m0 + wm * 32 + tm * 16 + r + half * 8;
                float v = acc[tm][tn][r] + bias[mrow];
                Yb[(size_t)mrow * SEQ_N + ncol] = (TOUT)v;
            }
        }
    }
}

// ---------------------------------------------------------------------------
// Flash attention per (b, h, 128-query block). K/V streamed in 128-wide
// chunks. Q pre-scaled by log2(e)/sqrt(64) so softmax uses exp2.
// V tile loaded by the Tensor Data Mover (wave 0 issues, s_wait_tensorcnt).
// Layouts: QsT[n][dd] (A of S), KsT[m][dd] (B of S),
//          Vs[dd][m] (B of PV, TDM-padded rows), Ps[n][m] (A of PV).
// ---------------------------------------------------------------------------
__global__ __launch_bounds__(256)
void attn_kernel(const _Float16* __restrict__ Qg, const _Float16* __restrict__ Kg,
                 const _Float16* __restrict__ Vg, _Float16* __restrict__ Xg)
{
    constexpr int BQ = 128, BKV = 128, LDD = HEAD_DIM + 8, LDM = BKV + 8;
    __shared__ __align__(16) _Float16 QsT[BQ][LDD];
    __shared__ __align__(16) _Float16 KsT[BKV][LDD];
    __shared__ __align__(16) _Float16 Vs[HEAD_DIM][LDM];
    __shared__ __align__(16) _Float16 Ps[BQ][LDM];

    const int tid   = threadIdx.x;
    const int lane  = tid & 31;
    const int wave  = tid >> 5;
    const int col16 = lane & 15;
    const int half  = lane >> 4;

    const int b  = blockIdx.z;
    const int h  = blockIdx.y;
    const int n0 = blockIdx.x * BQ;

    const size_t headBase = ((size_t)b * D_MODEL + h) * SEQ_N;

    // Q tile (64dd x 128n) -> QsT[n][dd], scaled by log2(e)/sqrt(HEAD_DIM)
    const _Float16 qscale = (_Float16)(0.125f * 1.44269504089f);
#pragma unroll
    for (int it = 0; it < 4; ++it) {
        int v  = it * 256 + tid;
        int dd = v >> 4;
        int p8 = (v & 15) * 8;
        v8h_t q = *(const v8h_t*)(Qg + headBase + (size_t)dd * 8 * SEQ_N + n0 + p8);
#pragma unroll
        for (int e = 0; e < 8; ++e)
            QsT[p8 + e][dd] = q[e] * qscale;
    }

    float mrow[8], lrow[8];
    v8f   ot[4];
#pragma unroll
    for (int r = 0; r < 8; ++r) { mrow[r] = -1e30f; lrow[r] = 0.0f; }
#pragma unroll
    for (int t = 0; t < 4; ++t)
#pragma unroll
        for (int e = 0; e < 8; ++e) ot[t][e] = 0.0f;

    for (int m0 = 0; m0 < SEQ_N; m0 += BKV) {
        // Issue TDM for the V tile: 64 rows x 128 f16, row stride 8*SEQ_N,
        // LDS rows padded: 64 DWORDs (code 5) then +4 DWORDs (code 3)
        if (wave == 0) {
            tdm_load_2d_f16((unsigned)(size_t)&Vs[0][0],
                            Vg + headBase + m0,
                            SEQ_N, HEAD_DIM, BKV, HEAD_DIM,
                            (unsigned long long)8 * SEQ_N,
                            /*pad_interval(64 dw)=*/5, /*pad_amount(4 dw)=*/3);
        }
        // Meanwhile all waves stage K transposed -> KsT[m][dd]
#pragma unroll
        for (int it = 0; it < 4; ++it) {
            int v  = it * 256 + tid;
            int dd = v >> 4;
            int p8 = (v & 15) * 8;
            v8h_t kk = *(const v8h_t*)(Kg + headBase + (size_t)dd * 8 * SEQ_N + m0 + p8);
#pragma unroll
            for (int e = 0; e < 8; ++e) KsT[p8 + e][dd] = kk[e];
        }
        if (wave == 0) __builtin_amdgcn_s_wait_tensorcnt(0);
        __syncthreads();

        // S strip (16n x 128m): 8 tiles, K=64 over dd as 2 WMMA steps
        v8f st[8];
#pragma unroll
        for (int t = 0; t < 8; ++t)
#pragma unroll
            for (int e = 0; e < 8; ++e) st[t][e] = 0.0f;

#pragma unroll
        for (int ks = 0; ks < 2; ++ks) {
            v16h a = load_a_frag(&QsT[wave * 16 + col16][0], ks * 32, half);
#pragma unroll
            for (int t = 0; t < 8; ++t) {
                v16h bb = load_b_frag(&KsT[t * 16 + col16][0], ks * 32, half);
                st[t] = __builtin_amdgcn_wmma_f32_16x16x32_f16(
                    false, a, false, bb, (short)0, st[t], false, false);
            }
        }

        // Online softmax in log2 domain; rows (r + half*8)
        float alpha[8];
#pragma unroll
        for (int r = 0; r < 8; ++r) {
            float mx = st[0][r];
#pragma unroll
            for (int t = 1; t < 8; ++t) mx = fmaxf(mx, st[t][r]);
#pragma unroll
            for (int off = 1; off < 16; off <<= 1)
                mx = fmaxf(mx, __shfl_xor(mx, off, 32));
            float mnew = fmaxf(mrow[r], mx);
            alpha[r]   = exp2f(mrow[r] - mnew);
            mrow[r]    = mnew;
            float s = 0.0f;
#pragma unroll
            for (int t = 0; t < 8; ++t) {
                float p = exp2f(st[t][r] - mnew);
                st[t][r] = p;
                s += p;
            }
#pragma unroll
            for (int off = 1; off < 16; off <<= 1)
                s += __shfl_xor(s, off, 32);
            lrow[r] = lrow[r] * alpha[r] + s;
        }

#pragma unroll
        for (int t = 0; t < 4; ++t)
#pragma unroll
            for (int r = 0; r < 8; ++r) ot[t][r] *= alpha[r];

        // Spill P (C-layout scatter) to wave-private strip; reloads are b128
#pragma unroll
        for (int t = 0; t < 8; ++t)
#pragma unroll
            for (int r = 0; r < 8; ++r)
                Ps[wave * 16 + r + half * 8][t * 16 + col16] = (_Float16)st[t][r];

        // O += P * V^T: K=128 over m as 4 steps, 4 dd tiles
#pragma unroll
        for (int ms = 0; ms < 4; ++ms) {
            v16h a = load_a_frag(&Ps[wave * 16 + col16][0], ms * 32, half);
#pragma unroll
            for (int t = 0; t < 4; ++t) {
                v16h bb = load_b_frag(&Vs[t * 16 + col16][0], ms * 32, half);
                ot[t] = __builtin_amdgcn_wmma_f32_16x16x32_f16(
                    false, a, false, bb, (short)0, ot[t], false, false);
            }
        }
        __syncthreads();
    }

    // Normalize, write x[b][dd*8+h][n] f16
#pragma unroll
    for (int t = 0; t < 4; ++t) {
        const int dd = t * 16 + col16;
        _Float16* dst = Xg + headBase + (size_t)dd * 8 * SEQ_N;
#pragma unroll
        for (int r = 0; r < 8; ++r) {
            const int n = n0 + wave * 16 + r + half * 8;
            dst[n] = (_Float16)(ot[t][r] / lrow[r]);
        }
    }
}

// ---------------------------------------------------------------------------
extern "C" void kernel_launch(void* const* d_in, const int* in_sizes, int n_in,
                              void* d_out, int out_size, void* d_ws, size_t ws_size,
                              hipStream_t stream) {
    const float* query = (const float*)d_in[0];
    const float* key   = (const float*)d_in[1];
    const float* value = (const float*)d_in[2];
    const float* Wq    = (const float*)d_in[3];
    const float* bq    = (const float*)d_in[4];
    const float* Wk    = (const float*)d_in[5];
    const float* bk    = (const float*)d_in[6];
    const float* Wv    = (const float*)d_in[7];
    const float* bv    = (const float*)d_in[8];
    const float* Wm    = (const float*)d_in[9];
    const float* bm    = (const float*)d_in[10];
    float* out = (float*)d_out;

    const size_t ELEMS = (size_t)BATCH * D_MODEL * SEQ_N;
    _Float16* Qh = (_Float16*)d_ws;
    _Float16* Kh = Qh + ELEMS;
    _Float16* Vh = Kh + ELEMS;
    _Float16* Xh = Vh + ELEMS;

    dim3 blk(256);
    dim3 gGemm(SEQ_N / 64, D_MODEL / 128, BATCH);
    dim3 gAttn(SEQ_N / 128, NUM_HEADS, BATCH);

    gemm_proj<float, _Float16><<<gGemm, blk, 0, stream>>>(Wq, bq, query, Qh);
    gemm_proj<float, _Float16><<<gGemm, blk, 0, stream>>>(Wk, bk, key,   Kh);
    gemm_proj<float, _Float16><<<gGemm, blk, 0, stream>>>(Wv, bv, value, Vh);
    attn_kernel<<<gAttn, blk, 0, stream>>>(Qh, Kh, Vh, Xh);
    gemm_proj<_Float16, float><<<gGemm, blk, 0, stream>>>(Wm, bm, Xh, out);
}
